// LengthRegulator_2001454760510
// MI455X (gfx1250) — matrix operand order repaired and checked by reading, not compiled, not run on hardware
//
#include <hip/hip_runtime.h>

typedef _Float16 half_t;
typedef __attribute__((ext_vector_type(16))) _Float16 v16h;
typedef __attribute__((ext_vector_type(8)))  _Float16 v8h;
typedef __attribute__((ext_vector_type(8)))  float    v8f;

#define B_SZ   32
#define L_SZ   512
#define D_SZ   384
#define F_SZ   384
#define KW     3
#define M_MEL  4096
#define ROWS   (B_SZ * L_SZ)     // 16384
#define KDIM   (D_SZ * KW)       // 1152
#define KSTEPS (KDIM / 32)       // 36
#define KPT    (D_SZ / 32)       // 12 K-steps per conv tap
#define NT_STRIDE (KSTEPS * 32 * 16)  // 18432 halfs between column tiles
#define KS_STRIDE (32 * 16)           // 512 halfs between K-steps
#define MROWS  32                // rows per block (2 M-tiles)

// ---------------------------------------------------------------------------
// f32 -> f16 convert
// ---------------------------------------------------------------------------
__global__ void cvt_f16_kernel(const float* __restrict__ src,
                               half_t* __restrict__ dst, int n) {
  int i = blockIdx.x * blockDim.x + threadIdx.x;
  if (i < n) dst[i] = (half_t)src[i];
}

// Zero-filled pad row (branchless conv padding target)
__global__ void zero_row_kernel(half_t* __restrict__ dst, int n) {
  int i = blockIdx.x * blockDim.x + threadIdx.x;
  if (i < n) dst[i] = (half_t)0.0f;
}

// ---------------------------------------------------------------------------
// Repack conv weights [F, Cin, K] (f32) into per-lane WMMA B-fragment layout:
// Bpack[((nt*KSTEPS + ks)*32 + lane)*16 + i] = B[k][n]  (f16)
//   n = nt*16 + (lane&15),  k = ks*32 + (lane>=16 ? 8:0) + (i&7) + (i>=8 ? 16:0) = kpos*D + d
//   B[k][n] = w[n][d][kpos];  16 contiguous halfs per lane -> two b128 loads.
// ---------------------------------------------------------------------------
__global__ void pack_w_kernel(const float* __restrict__ w,
                              half_t* __restrict__ bp) {
  int tid = blockIdx.x * blockDim.x + threadIdx.x;
  if (tid >= KDIM * F_SZ) return;
  int i    = tid & 15;
  int r    = tid >> 4;
  int lane = r & 31;
  int r2   = r >> 5;
  int ks   = r2 % KSTEPS;
  int nt   = r2 / KSTEPS;
  int n    = nt * 16 + (lane & 15);
  int K    = ks * 32 + ((lane & 16) ? 8 : 0) + (i & 7) + ((i & 8) ? 16 : 0);
  int kpos = K / D_SZ;
  int d    = K - kpos * D_SZ;
  bp[tid]  = (half_t)w[(n * D_SZ + d) * KW + kpos];
}

// ---------------------------------------------------------------------------
// Fused conv-as-GEMM (WMMA f16->f32) + bias + LayerNorm + ReLU.
// Block = 32 rows x full F=384. 6 waves; each wave owns a 2(M) x 4(N) grid of
// 16x16 accumulator tiles, so every B fragment pair feeds TWO v_wmma ops
// (load:wmma = 1.5:1). Branchless padding (zeroed row), all fragment loads
// are immediate offsets from hoisted base pointers.
// Mode A (Hout): write f16 normalized activations. Mode B (dur): fuse the
// final [F,1] linear + ReLU via LDS float atomics.
// ---------------------------------------------------------------------------
__global__ void __launch_bounds__(192)
gemm_ln_kernel(const half_t* __restrict__ A,     // [ROWS, D_SZ] f16
               const half_t* __restrict__ zrow,  // [D_SZ] zeros (pad row)
               const half_t* __restrict__ Bp,    // packed weights
               const float* __restrict__ cbias,  // [F]
               const float* __restrict__ g,      // [F] ln gamma
               const float* __restrict__ beta,   // [F] ln beta
               half_t* __restrict__ Hout,        // [ROWS, F] or null
               const float* __restrict__ linw,   // [F] or null
               const float* __restrict__ linb,   // [1] or null
               float* __restrict__ dur)          // [ROWS] or null
{
  __shared__ float tile[MROWS][392];
  __shared__ float psum[MROWS][6];
  __shared__ float psq[MROWS][6];
  __shared__ float mu[MROWS], rsg[MROWS];
  __shared__ float dacc[MROWS];

  const int tile_m = blockIdx.x;                 // 32-row tile index
  const int t      = threadIdx.x;
  const int wave   = t >> 5;
  const int lane   = t & 31;
  const int mlane  = lane & 15;
  const int hi     = lane >> 4;       // A: 0 -> K{0..7,16..23}, 1 -> K{8..15,24..31}
  // Two A-matrix rows per lane (M-tile 0 and 1); 32-row tile never crosses a
  // batch boundary (512 % 32 == 0).
  const int row0   = tile_m * MROWS + mlane;
  const int row1   = row0 + 16;
  const int l0     = row0 & (L_SZ - 1);
  const int l1     = row1 & (L_SZ - 1);
  const int bidx   = row0 >> 9;

  v8f acc0[4] = {};
  v8f acc1[4] = {};

  // Per-wave B base (column tiles nt = wave*4 + j)
  const half_t* bbase = Bp + (size_t)(wave * 4) * NT_STRIDE + lane * 16;

  for (int tap = 0; tap < KW; ++tap) {
    const int s0 = l0 + tap - 1;                  // pad = 1
    const int s1 = l1 + tap - 1;
    const half_t* ap0 = (((unsigned)s0 < (unsigned)L_SZ)
                           ? (A + (size_t)(bidx * L_SZ + s0) * D_SZ) : zrow) + hi * 8;
    const half_t* ap1 = (((unsigned)s1 < (unsigned)L_SZ)
                           ? (A + (size_t)(bidx * L_SZ + s1) * D_SZ) : zrow) + hi * 8;
    const half_t* bp = bbase + (size_t)tap * KPT * KS_STRIDE;

#pragma unroll
    for (int k = 0; k < KPT; ++k) {
      union { v16h v; v8h h[2]; } a0, a1;
      a0.h[0] = *(const v8h*)(ap0 + k * 32);
      a0.h[1] = *(const v8h*)(ap0 + k * 32 + 16);
      a1.h[0] = *(const v8h*)(ap1 + k * 32);
      a1.h[1] = *(const v8h*)(ap1 + k * 32 + 16);

      union { v16h v; v8h h[2]; } bfr[4];
#pragma unroll
      for (int j = 0; j < 4; ++j) {
        const half_t* bj = bp + (size_t)j * NT_STRIDE + k * KS_STRIDE;
        bfr[j].h[0] = *(const v8h*)(bj);
        bfr[j].h[1] = *(const v8h*)(bj + 8);
      }
#pragma unroll
      for (int j = 0; j < 4; ++j) {
        acc0[j] = __builtin_amdgcn_wmma_f32_16x16x32_f16(
            false, a0.v, false, bfr[j].v, (short)0, acc0[j], false, false);
        acc1[j] = __builtin_amdgcn_wmma_f32_16x16x32_f16(
            false, a1.v, false, bfr[j].v, (short)0, acc1[j], false, false);
      }
    }
  }

  // Scatter C (+conv bias) into LDS. C layout: lane<16 -> M=p, N=lane;
  // lane>=16 -> M=8+p, N=lane-16.
#pragma unroll
  for (int j = 0; j < 4; ++j) {
    const int n = (wave * 4 + j) * 16 + mlane;
    const float cb = cbias[n];
#pragma unroll
    for (int p = 0; p < 8; ++p) {
      tile[p + hi * 8][n]      = acc0[j][p] + cb;
      tile[16 + p + hi * 8][n] = acc1[j][p] + cb;
    }
  }
  __syncthreads();

  // LayerNorm reduction: 192 threads = 32 rows x 6 chunks of 64.
  {
    const int prow  = t & 31;
    const int chunk = t >> 5;                 // 0..5
    float s = 0.f, s2 = 0.f;
    for (int c = 0; c < 64; ++c) {
      float v = tile[prow][chunk * 64 + c];
      s += v; s2 += v * v;
    }
    psum[prow][chunk] = s;
    psq [prow][chunk] = s2;
  }
  __syncthreads();
  if (t < MROWS) {
    float s = 0.f, s2 = 0.f;
    for (int c = 0; c < 6; ++c) { s += psum[t][c]; s2 += psq[t][c]; }
    float mean = s * (1.0f / (float)F_SZ);
    float var  = s2 * (1.0f / (float)F_SZ) - mean * mean;
    mu[t]  = mean;
    rsg[t] = rsqrtf(var + 1e-5f);
    dacc[t] = 0.f;
  }
  __syncthreads();

  // Normalize + ReLU; 12288 elements, 64 per thread, coalesced (stride 192).
  for (int it = 0; it < 64; ++it) {
    const int e = t + 192 * it;
    const int r = e / F_SZ;
    const int c = e - r * F_SZ;
    float v = (tile[r][c] - mu[r]) * rsg[r] * g[c] + beta[c];
    v = v > 0.f ? v : 0.f;
    if (Hout) {
      Hout[(size_t)(tile_m * MROWS + r) * F_SZ + c] = (half_t)v;
    } else {
      atomicAdd(&dacc[r], v * linw[c]);      // fused final linear (ds_add_f32)
    }
  }
  if (!Hout) {
    __syncthreads();
    if (t < MROWS) {
      float dv = dacc[t] + linb[0];
      dur[tile_m * MROWS + t] = dv > 0.f ? dv : 0.f;
    }
  }
}

// ---------------------------------------------------------------------------
// Per-batch inclusive cumsum of durations + searchsorted(right) index map.
// idxmap[b][t] = min(#cum <= t, 511) if t < total else -1
// ---------------------------------------------------------------------------
__global__ void __launch_bounds__(512)
scan_idx_kernel(const int* __restrict__ target, int* __restrict__ idxmap) {
  __shared__ int cum[512];
  __shared__ int totalS;
  const int b = blockIdx.x;
  const int t = threadIdx.x;
  cum[t] = target[b * L_SZ + t];
  __syncthreads();
  for (int off = 1; off < 512; off <<= 1) {
    int v = (t >= off) ? cum[t - off] : 0;
    __syncthreads();
    cum[t] += v;
    __syncthreads();
  }
  if (t == 511) totalS = cum[511];
  __syncthreads();
  const int total = totalS;
  for (int tt = t; tt < M_MEL; tt += 512) {
    int lo = 0, hiI = 512;
    while (lo < hiI) {                       // first i with cum[i] > tt
      int mid = (lo + hiI) >> 1;
      if (cum[mid] <= tt) lo = mid + 1; else hiI = mid;
    }
    idxmap[b * M_MEL + tt] = (tt < total) ? (lo < 511 ? lo : 511) : -1;
  }
}

// ---------------------------------------------------------------------------
// Length-regulator expand: out[b, t, :] = x[b, idx, :] or zeros. float4 rows.
// ---------------------------------------------------------------------------
__global__ void __launch_bounds__(96)
expand_kernel(const float* __restrict__ x, const int* __restrict__ idxmap,
              float* __restrict__ out) {
  const int b  = blockIdx.y;
  const int tt = blockIdx.x;
  const int c4 = threadIdx.x;                // 0..95 (384/4)
  const int idx = idxmap[b * M_MEL + tt];
  float4 v = make_float4(0.f, 0.f, 0.f, 0.f);
  if (idx >= 0)
    v = ((const float4*)x)[(size_t)(b * L_SZ + idx) * (D_SZ / 4) + c4];
  ((float4*)out)[(size_t)(b * M_MEL + tt) * (D_SZ / 4) + c4] = v;
}

// ---------------------------------------------------------------------------
extern "C" void kernel_launch(void* const* d_in, const int* in_sizes, int n_in,
                              void* d_out, int out_size, void* d_ws, size_t ws_size,
                              hipStream_t stream) {
  const float* x      = (const float*)d_in[0];
  const float* c1_w   = (const float*)d_in[1];
  const float* c1_b   = (const float*)d_in[2];
  const float* ln1_g  = (const float*)d_in[3];
  const float* ln1_b  = (const float*)d_in[4];
  const float* c2_w   = (const float*)d_in[5];
  const float* c2_b   = (const float*)d_in[6];
  const float* ln2_g  = (const float*)d_in[7];
  const float* ln2_b  = (const float*)d_in[8];
  const float* lin_w  = (const float*)d_in[9];
  const float* lin_b  = (const float*)d_in[10];
  const int*   target = (const int*)d_in[11];
  (void)in_sizes; (void)n_in; (void)out_size; (void)ws_size;

  char* ws = (char*)d_ws;
  size_t off = 0;
  auto walloc = [&](size_t bytes) -> void* {
    void* p = ws + off;
    off += (bytes + 255) & ~(size_t)255;
    return p;
  };
  half_t* xh   = (half_t*)walloc((size_t)ROWS * D_SZ * sizeof(half_t));
  half_t* h1h  = (half_t*)walloc((size_t)ROWS * F_SZ * sizeof(half_t));
  half_t* bp1  = (half_t*)walloc((size_t)KDIM * F_SZ * sizeof(half_t));
  half_t* bp2  = (half_t*)walloc((size_t)KDIM * F_SZ * sizeof(half_t));
  half_t* zrow = (half_t*)walloc((size_t)D_SZ * sizeof(half_t));
  int*    idxm = (int*)   walloc((size_t)B_SZ * M_MEL * sizeof(int));

  float* out = (float*)d_out;                         // [B, M, D]
  float* dur = out + (size_t)B_SZ * M_MEL * D_SZ;     // [B, L]

  {
    int n = ROWS * D_SZ;
    cvt_f16_kernel<<<(n + 255) / 256, 256, 0, stream>>>(x, xh, n);
  }
  zero_row_kernel<<<1, D_SZ, 0, stream>>>(zrow, D_SZ);
  {
    int n = KDIM * F_SZ;
    pack_w_kernel<<<(n + 255) / 256, 256, 0, stream>>>(c1_w, bp1);
    pack_w_kernel<<<(n + 255) / 256, 256, 0, stream>>>(c2_w, bp2);
  }
  // conv1 + LN1 + ReLU -> f16 activations
  gemm_ln_kernel<<<ROWS / MROWS, 192, 0, stream>>>(
      xh, zrow, bp1, c1_b, ln1_g, ln1_b, h1h, nullptr, nullptr, nullptr);
  // conv2 + LN2 + ReLU + linear + ReLU -> dur_pred
  gemm_ln_kernel<<<ROWS / MROWS, 192, 0, stream>>>(
      h1h, zrow, bp2, c2_b, ln2_g, ln2_b, nullptr, lin_w, lin_b, dur);
  // length regulation
  scan_idx_kernel<<<B_SZ, 512, 0, stream>>>(target, idxm);
  expand_kernel<<<dim3(M_MEL, B_SZ), 96, 0, stream>>>(x, idxm, out);
}
